// GATModel_42296837931390
// MI455X (gfx1250) — matrix operand order repaired and checked by reading, not compiled, not run on hardware
//
#include <hip/hip_runtime.h>
#include <math.h>

typedef __attribute__((ext_vector_type(16))) _Float16 v16h;
typedef __attribute__((ext_vector_type(8)))  _Float16 v8h;
typedef __attribute__((ext_vector_type(8)))  float    v8f;
typedef __attribute__((ext_vector_type(4)))  unsigned int u32x4;
typedef __attribute__((ext_vector_type(8)))  int          i32x8;
typedef __attribute__((ext_vector_type(4)))  int          i32x4;

#define BATCH 4
#define NNODE 512
#define DH    128
#define NH    8
#define NF    16
#define SLOPE 0.2f

// ---------------------------------------------------------------------------
// TDM: DMA one [512 x 16] f32 panel (row stride DH) from global into LDS.
// D# per CDNA5 ISA (2D tensor, groups 2/3 zero).
// This toolchain exposes the 6-arg builtin:
//   (u32x4 g0, i32x8 g1, i32x4 g2, i32x4 g3, i32x8 extra, i32 cpol)
// ---------------------------------------------------------------------------
__device__ __forceinline__ void tdm_load_panel(const float* gsrc,
                                               unsigned lds_off) {
  const unsigned long long ga = (unsigned long long)(uintptr_t)gsrc;
  u32x4 g0;
  g0[0] = 1u;                                            // count=1, user mode
  g0[1] = lds_off;                                       // lds_addr [63:32]
  g0[2] = (unsigned)ga;                                  // global_addr lo
  g0[3] = ((unsigned)(ga >> 32) & 0x01FFFFFFu)           // global_addr hi
          | (2u << 30);                                  // type=2 ("image")
  i32x8 g1;
  g1[0] = 0x00020000;                    // workgroup_mask=0, data_size=4B
  g1[1] = (int)(16u << 16);              // tensor_dim0 = 16   [79:48]
  g1[2] = (int)((unsigned)NNODE << 16);  // tensor_dim1 = 512  [111:80]
  g1[3] = (int)(16u << 16);              // tile_dim0 = 16     [127:112]
  g1[4] = NNODE;                         // tile_dim1 = 512, tile_dim2 = 0
  g1[5] = DH;                            // tensor_dim0_stride = 128 (lo32)
  g1[6] = 0;                             // stride0 hi, stride1 lo16
  g1[7] = 1;                             // tensor_dim1_stride = 65536 (hi32)
  const i32x4 z4 = {0, 0, 0, 0};         // groups 2/3: 2D tensor -> unused
  const i32x8 z8 = {0, 0, 0, 0, 0, 0, 0, 0};
  __builtin_amdgcn_tensor_load_to_lds(g0, g1, z4, z4, z8, 0);
}

__device__ __forceinline__ unsigned lds_offset_of(const void* p) {
  return (unsigned)(unsigned long long)
      (__attribute__((address_space(3))) const char*)p;
}

// ---------------------------------------------------------------------------
// h = X @ W_in   (Din = 2, trivial)
// ---------------------------------------------------------------------------
__global__ void gat_in_proj(const float* __restrict__ x,
                            const float* __restrict__ Win,
                            float* __restrict__ h) {
  const int idx = blockIdx.x * blockDim.x + threadIdx.x;
  if (idx >= BATCH * NNODE * DH) return;
  const int row = idx / DH, d = idx % DH;
  h[idx] = fmaf(x[row * 2 + 0], Win[d], x[row * 2 + 1] * Win[DH + d]);
}

// ---------------------------------------------------------------------------
// g_l = h @ Wl ; g_r = h @ Wr    ([2048x128]x[128x128], f16 WMMA, f32 acc)
// One wave computes one 16x16 tile of BOTH outputs (A fragment reused).
// ---------------------------------------------------------------------------
__global__ __launch_bounds__(32)
void gat_proj_kernel(const float* __restrict__ h,
                     const float* __restrict__ Wl,
                     const float* __restrict__ Wr,
                     float* __restrict__ gl,
                     float* __restrict__ gr) {
  const int lane = threadIdx.x;
  const int m0 = blockIdx.x * 16;   // row tile over B*N
  const int n0 = blockIdx.y * 16;   // col tile over DH
  v8f cl = {}, cr = {};
  const int am  = m0 + (lane & 15);
  const int bn  = n0 + (lane & 15);
  const int off = (lane & 16) >> 1;            // 0 / 8: K sub-offset per half

#pragma unroll
  for (int kc = 0; kc < DH; kc += 32) {
    // A fragment: K = kc+off..+7 and kc+16+off..+7  (two 32B runs)
    float tmp[16];
    const float4* hp = (const float4*)(h + am * DH + kc + off);
    const float4* hq = (const float4*)(h + am * DH + kc + 16 + off);
    *(float4*)&tmp[0]  = hp[0];
    *(float4*)&tmp[4]  = hp[1];
    *(float4*)&tmp[8]  = hq[0];
    *(float4*)&tmp[12] = hq[1];
    v16h a, bl, br;
#pragma unroll
    for (int p = 0; p < 16; ++p) {
      a[p] = (_Float16)tmp[p];
      const int kb = kc + (p < 8 ? p + off : p + 8 + off);   // B: K = off+p, 16+off+p-8
      bl[p] = (_Float16)Wl[kb * DH + bn];
      br[p] = (_Float16)Wr[kb * DH + bn];
    }
    cl = __builtin_amdgcn_wmma_f32_16x16x32_f16(false, a, false, bl,
                                                (short)0, cl, false, false);
    cr = __builtin_amdgcn_wmma_f32_16x16x32_f16(false, a, false, br,
                                                (short)0, cr, false, false);
  }
#pragma unroll
  for (int r = 0; r < 8; ++r) {                // C layout: M=r(+8), N=lane
    const int m = m0 + r + ((lane >> 4) << 3);
    const int n = n0 + (lane & 15);
    gl[m * DH + n] = cl[r];
    gr[m * DH + n] = cr[r];
  }
}

// ---------------------------------------------------------------------------
// Fused scores + masked softmax + aggregation for one (batch, head, 16-row
// tile). Single wave per workgroup (EXEC all-ones for WMMA). Panels are DMA'd
// into LDS by the Tensor Data Mover; aggregation is 16 unrolled f16 WMMAs.
// LDS: 32KB scores + 32KB raw f32 panel + 16KB f16 transposed panel = 80KB.
// ---------------------------------------------------------------------------
__global__ __launch_bounds__(32)
void gat_attn_kernel(const float* __restrict__ gl,
                     const float* __restrict__ gr,
                     const int* __restrict__ adj,
                     const float* __restrict__ attn_a,
                     float* __restrict__ hout) {
  __shared__ __align__(16) float    e_lds[16 * NNODE];    // scores / probs
  __shared__ __align__(16) float    rawp[NNODE * NF];     // TDM panel [j][f]
  __shared__ __align__(16) _Float16 t_lds[NF * NNODE];    // B operand [f][j]

  const int lane  = threadIdx.x;
  const int itile = blockIdx.x & 31;
  const int head  = (blockIdx.x >> 5) & 7;
  const int b     = blockIdx.x >> 8;
  const int i0    = itile * 16;
  const int gbase = b * NNODE * DH + head * NF;
  const unsigned rawp_off = lds_offset_of(rawp);

  // kick off TDM: g_l[b, :, head, :] panel -> LDS (f32, row-major)
  tdm_load_panel(gl + gbase, rawp_off);

  float av[NF];
#pragma unroll
  for (int f = 0; f < NF; ++f) av[f] = attn_a[f];

  // this lane's destination row (2 lanes per row, split j-range)
  const int i    = lane & 15;
  const int ig   = i0 + i;
  const int half = lane >> 4;
  float gri[NF];
#pragma unroll
  for (int f = 0; f < NF; ++f) gri[f] = gr[gbase + ig * DH + f];

  const int* adj_row = adj + (b * NNODE + ig) * NNODE;

  __builtin_amdgcn_s_wait_tensorcnt(0);
  __syncthreads();   // single-wave WG: NOP barrier, flushes LDS ordering

  // e[i][j] = a . LeakyReLU(g_r[i] + g_l[j]) , adjacency-masked
  for (int j = half; j < NNODE; j += 2) {
    float e = 0.f;
#pragma unroll
    for (int f = 0; f < NF; ++f) {
      float s = gri[f] + rawp[j * NF + f];
      s = (s > 0.f) ? s : SLOPE * s;
      e = fmaf(av[f], s, e);
    }
    e_lds[i * NNODE + j] = (adj_row[j] != 0) ? e : -__builtin_inff();
  }

  // softmax over j: each half-lane scans its stride, combine via shfl_xor(16)
  float m = -__builtin_inff();
  for (int j = half; j < NNODE; j += 2) m = fmaxf(m, e_lds[i * NNODE + j]);
  m = fmaxf(m, __shfl_xor(m, 16, 32));
  float s = 0.f;
  for (int j = half; j < NNODE; j += 2) s += __expf(e_lds[i * NNODE + j] - m);
  s += __shfl_xor(s, 16, 32);
  const float rinv = 1.0f / s;

  for (int j = half; j < NNODE; j += 2)
    e_lds[i * NNODE + j] = __expf(e_lds[i * NNODE + j] - m) * rinv;

  __syncthreads();   // all rawp readers done before TDM overwrites it

  // TDM: g_r[b, :, head, :] panel -> LDS, then transpose/convert to f16
  tdm_load_panel(gr + gbase, rawp_off);
  __builtin_amdgcn_s_wait_tensorcnt(0);
  __syncthreads();

  for (int t = lane; t < NNODE * NF; t += 32) {
    const int j = t >> 4, f = t & 15;
    t_lds[f * NNODE + j] = (_Float16)rawp[t];
  }
  __syncthreads();

  // out[i, f] = sum_j p[i,j] * g_r[j, f]  -> 16 unrolled WMMAs over K=512
  const int off = (lane & 16) >> 1;            // 0 / 8
  v8f c = {};
#pragma unroll
  for (int kc = 0; kc < NNODE; kc += 32) {
    // A fragment from probs: two contiguous 32B runs -> ds_load_b128 x4
    float tmp[16];
    const float4* ep = (const float4*)&e_lds[i * NNODE + kc + off];
    const float4* eq = (const float4*)&e_lds[i * NNODE + kc + 16 + off];
    *(float4*)&tmp[0]  = ep[0];
    *(float4*)&tmp[4]  = ep[1];
    *(float4*)&tmp[8]  = eq[0];
    *(float4*)&tmp[12] = eq[1];
    v16h a;
#pragma unroll
    for (int p = 0; p < 16; ++p) a[p] = (_Float16)tmp[p];

    // B fragment: lane n, K runs kc+off..+7 and kc+16+off..+7 in transposed panel
    const v8h* bp = (const v8h*)&t_lds[(lane & 15) * NNODE + kc + off];
    const v8h b0 = bp[0], b1 = bp[2];
    v16h bm;
#pragma unroll
    for (int p = 0; p < 8; ++p) { bm[p] = b0[p]; bm[p + 8] = b1[p]; }

    c = __builtin_amdgcn_wmma_f32_16x16x32_f16(false, a, false, bm,
                                               (short)0, c, false, false);
  }

#pragma unroll
  for (int r = 0; r < 8; ++r) {
    const int mrow = i0 + r + ((lane >> 4) << 3);
    hout[b * NNODE * DH + mrow * DH + head * NF + (lane & 15)] = c[r];
  }
}

// ---------------------------------------------------------------------------
// out = h @ W_out   (Dout = 1), vectorized row dot-product
// ---------------------------------------------------------------------------
__global__ void gat_out_proj(const float* __restrict__ h,
                             const float* __restrict__ Wout,
                             float* __restrict__ out) {
  const int idx = blockIdx.x * blockDim.x + threadIdx.x;
  if (idx >= BATCH * NNODE) return;
  const float4* hp = (const float4*)(h + idx * DH);
  const float4* wp = (const float4*)Wout;
  float s = 0.f;
#pragma unroll 8
  for (int d = 0; d < DH / 4; ++d) {
    const float4 hv = hp[d], wv = wp[d];
    s = fmaf(hv.x, wv.x, s);
    s = fmaf(hv.y, wv.y, s);
    s = fmaf(hv.z, wv.z, s);
    s = fmaf(hv.w, wv.w, s);
  }
  out[idx] = s;
}

// ---------------------------------------------------------------------------
extern "C" void kernel_launch(void* const* d_in, const int* in_sizes, int n_in,
                              void* d_out, int out_size, void* d_ws, size_t ws_size,
                              hipStream_t stream) {
  (void)in_sizes; (void)n_in; (void)out_size; (void)ws_size;
  const float* x   = (const float*)d_in[0];
  const int*   adj = (const int*)d_in[1];
  const float* Win = (const float*)d_in[2];
  const float* Wl  = (const float*)d_in[3];
  const float* Wr  = (const float*)d_in[4];
  const float* aa  = (const float*)d_in[5];
  const float* Wo  = (const float*)d_in[6];
  float* out = (float*)d_out;

  char* ws = (char*)d_ws;
  const size_t HB = (size_t)BATCH * NNODE * DH * sizeof(float);  // 1 MB each
  float* h0 = (float*)(ws);
  float* h1 = (float*)(ws + HB);
  float* glb = (float*)(ws + 2 * HB);
  float* grb = (float*)(ws + 3 * HB);

  gat_in_proj<<<(BATCH * NNODE * DH + 255) / 256, 256, 0, stream>>>(x, Win, h0);

  float* hc = h0;
  float* hn = h1;
  for (int l = 0; l < 3; ++l) {
    dim3 pg(BATCH * NNODE / 16, DH / 16);   // 128 x 8 tiles
    gat_proj_kernel<<<pg, 32, 0, stream>>>(hc, Wl + (size_t)l * DH * DH,
                                           Wr + (size_t)l * DH * DH, glb, grb);
    gat_attn_kernel<<<BATCH * NH * (NNODE / 16), 32, 0, stream>>>(
        glb, grb, adj, aa + l * NF, hn);
    float* t = hc; hc = hn; hn = t;
  }

  gat_out_proj<<<(BATCH * NNODE + 255) / 256, 256, 0, stream>>>(hc, Wo, out);
}
